// SageDy_31507880083676
// MI455X (gfx1250) — matrix operand order repaired and checked by reading, not compiled, not run on hardware
//
#include <hip/hip_runtime.h>
#include <hip/hip_bf16.h>

// ---------------------------------------------------------------------------
// GraphSAGE 2-layer forward for MI455X (gfx1250, wave32, WMMA)
//   gathers + neighbor means : HBM-bound, float4 loads, bf16x4 packed stores
//   dense transforms         : bf16 WMMA, one wave = 16x128 row-block,
//                              8 accumulators, A reused across 8 N-tiles,
//                              B fragments batched x4 (no spill, partial waits)
// ---------------------------------------------------------------------------

typedef __attribute__((ext_vector_type(16))) __bf16 v16bf;
typedef __attribute__((ext_vector_type(8)))  __bf16 v8bf;
typedef __attribute__((ext_vector_type(4)))  __bf16 v4bf;
typedef __attribute__((ext_vector_type(8)))  float  v8f;
typedef __attribute__((ext_vector_type(4)))  float  v4f;

#define D128 128

// ---------------- Phase-1: gather / gather-mean (f32 -> bf16) --------------
// 128-thread blocks: 4 rows x 32 lanes; each lane moves 4 columns (16B load).

__global__ void sage_gather_bf16(const float* __restrict__ feat,
                                 const int* __restrict__ idx,
                                 __bf16* __restrict__ out) {
  const int r = blockIdx.x * 4 + (threadIdx.x >> 5);
  const int c = (threadIdx.x & 31) * 4;
  const v4f x = *(const v4f*)(feat + (long long)idx[r] * D128 + c);
  *(v4bf*)(out + (long long)r * D128 + c) = __builtin_convertvector(x, v4bf);
}

template <int FAN>
__global__ void sage_gather_mean_bf16(const float* __restrict__ feat,
                                      const int* __restrict__ idx,
                                      __bf16* __restrict__ out, float inv) {
  const int r = blockIdx.x * 4 + (threadIdx.x >> 5);
  const int c = (threadIdx.x & 31) * 4;
  const int* ip = idx + (long long)r * FAN;
  v4f s = {0.f, 0.f, 0.f, 0.f};
#pragma unroll
  for (int j = 0; j < FAN; ++j) {
    s += *(const v4f*)(feat + (long long)ip[j] * D128 + c);
  }
  s *= inv;
  *(v4bf*)(out + (long long)r * D128 + c) = __builtin_convertvector(s, v4bf);
}

template <int FAN>
__global__ void sage_mean_rows_bf16(const __bf16* __restrict__ in,
                                    __bf16* __restrict__ out, float inv) {
  const int r = blockIdx.x * 4 + (threadIdx.x >> 5);
  const int c = (threadIdx.x & 31) * 4;
  v4f s = {0.f, 0.f, 0.f, 0.f};
#pragma unroll
  for (int j = 0; j < FAN; ++j) {
    const v4bf x = *(const v4bf*)(in + ((long long)r * FAN + j) * D128 + c);
    s += __builtin_convertvector(x, v4f);
  }
  s *= inv;
  *(v4bf*)(out + (long long)r * D128 + c) = __builtin_convertvector(s, v4bf);
}

// ---------------- Weight pack: f32 [128x128] -> bf16 fragment layout -------
// Wp[kc(4)][nTile(8)][lane(32)][e(16)] ; lane n<16 -> col nT*16+n, K=kc*32+{0..7,16..23}[e]
//                                        lane n>=16 -> same col, K=kc*32+{8..15,24..31}[e]
__global__ void sage_pack_w(const float* __restrict__ W,
                            __bf16* __restrict__ Wp) {
  const int t    = blockIdx.x * 256 + threadIdx.x;   // 16384 total
  const int e    = t & 15;
  const int lane = (t >> 4) & 31;
  const int nT   = (t >> 9) & 7;
  const int kc   = t >> 12;
  const int col  = nT * 16 + (lane & 15);
  const int k    = kc * 32 + ((lane >> 4) << 3) + (e & 7) + ((e >> 3) << 4);
  Wp[t] = (__bf16)W[k * D128 + col];
}

// ---------------- Phase-2: fused dual-GEMM via WMMA ------------------------
// Y[M x 128] = act( A1 @ W1 + A2 @ W2 )
// One wave per 16x128 row-block: 8 f32 accumulators, 64 WMMAs.
// Per k-chunk: 1 A fragment; B fragments in two batches of 4 (8x b128
// clause each) so WMMAs drain with partial loadcnt waits, no VGPR spill.

// A fragment (16x32 bf16), ISA 7.12.2 layout:
//  lanes 0-15 : M = lane,    K = {k0+0..7,  k0+16..23}
//  lanes16-31 : M = lane-16, K = {k0+8..15, k0+24..31}
__device__ inline v16bf load_frag_a(const __bf16* __restrict__ A,
                                    int m0, int k0, int lane) {
  const int row = m0 + (lane & 15);
  const int kb  = k0 + ((lane >> 4) << 3);
  const __bf16* p = A + (long long)row * D128 + kb;
  const v8bf lo = *(const v8bf*)(p);        // 16B aligned
  const v8bf hi = *(const v8bf*)(p + 16);
  return __builtin_shufflevector(lo, hi, 0, 1, 2, 3, 4, 5, 6, 7,
                                 8, 9, 10, 11, 12, 13, 14, 15);
}

// B fragment from packed weights: one lane's 16 values are contiguous (32B).
__device__ inline v16bf load_frag_b(const __bf16* __restrict__ Wp,
                                    int kc, int nT, int lane) {
  const __bf16* p = Wp + ((((kc << 3) + nT) << 5) + lane) * 16;
  const v8bf lo = *(const v8bf*)(p);
  const v8bf hi = *(const v8bf*)(p + 8);
  return __builtin_shufflevector(lo, hi, 0, 1, 2, 3, 4, 5, 6, 7,
                                 8, 9, 10, 11, 12, 13, 14, 15);
}

template <bool RELU, bool TOF32>
__global__ void sage_dual_gemm_wmma(const __bf16* __restrict__ A1,
                                    const __bf16* __restrict__ A2,
                                    const __bf16* __restrict__ W1p,
                                    const __bf16* __restrict__ W2p,
                                    __bf16* __restrict__ outB,
                                    float* __restrict__ outF) {
  const int lane = threadIdx.x & 31;
  const int wave = threadIdx.x >> 5;
  const int m0 = (blockIdx.x * 8 + wave) << 4;   // this wave's 16-row block

  v8f acc[8] = {};   // 8 N-tiles of 16x16 f32

#pragma unroll
  for (int kc = 0; kc < 4; ++kc) {               // K = 128 over A1/W1
    const v16bf a = load_frag_a(A1, m0, kc * 32, lane);
#pragma unroll
    for (int g = 0; g < 2; ++g) {                // two batches of 4 N-tiles
      v16bf b[4];
#pragma unroll
      for (int i = 0; i < 4; ++i) b[i] = load_frag_b(W1p, kc, g * 4 + i, lane);
#pragma unroll
      for (int i = 0; i < 4; ++i) {
        acc[g * 4 + i] = __builtin_amdgcn_wmma_f32_16x16x32_bf16(
            false, a, false, b[i], (short)0, acc[g * 4 + i], false, false);
      }
    }
  }
#pragma unroll
  for (int kc = 0; kc < 4; ++kc) {               // K = 128 over A2/W2
    const v16bf a = load_frag_a(A2, m0, kc * 32, lane);
#pragma unroll
    for (int g = 0; g < 2; ++g) {
      v16bf b[4];
#pragma unroll
      for (int i = 0; i < 4; ++i) b[i] = load_frag_b(W2p, kc, g * 4 + i, lane);
#pragma unroll
      for (int i = 0; i < 4; ++i) {
        acc[g * 4 + i] = __builtin_amdgcn_wmma_f32_16x16x32_bf16(
            false, a, false, b[i], (short)0, acc[g * 4 + i], false, false);
      }
    }
  }

  // D layout: VGPR r, lane l -> M = m0 + r + 8*(l>=16), N = nT*16 + (l&15)
  const int colLo = lane & 15;
  const int rbase = m0 + ((lane >> 4) << 3);
#pragma unroll
  for (int nT = 0; nT < 8; ++nT) {
    const int col = (nT << 4) + colLo;
#pragma unroll
    for (int r = 0; r < 8; ++r) {
      float v = acc[nT][r];
      if (RELU) v = fmaxf(v, 0.f);
      const long long o = (long long)(rbase + r) * D128 + col;
      if (TOF32) outF[o] = v;
      else       outB[o] = (__bf16)v;
    }
  }
}

// ---------------------------------------------------------------------------

extern "C" void kernel_launch(void* const* d_in, const int* in_sizes, int n_in,
                              void* d_out, int out_size, void* d_ws, size_t ws_size,
                              hipStream_t stream) {
  (void)in_sizes; (void)n_in; (void)out_size; (void)ws_size;

  const float* features = (const float*)d_in[0];
  const float* W_self_0 = (const float*)d_in[1];
  const float* W_neigh_0 = (const float*)d_in[2];
  const float* W_self_1 = (const float*)d_in[3];
  const float* W_neigh_1 = (const float*)d_in[4];
  const int* samples0 = (const int*)d_in[5];
  const int* samples1 = (const int*)d_in[6];
  const int* samples2 = (const int*)d_in[7];
  float* out = (float*)d_out;

  const int B  = 1024;
  const int S2 = 10;
  const int WN = D128 * D128;   // 16384 elements per weight matrix

  // ---- carve bf16 scratch from d_ws ----
  char* base = (char*)d_ws;
  size_t off = 0;
  auto alloc = [&](size_t elems) {
    __bf16* p = (__bf16*)(base + off);
    off += ((elems * sizeof(__bf16)) + 255) & ~size_t(255);
    return p;
  };
  __bf16* Ws0p  = alloc(WN);
  __bf16* Wn0p  = alloc(WN);
  __bf16* Ws1p  = alloc(WN);
  __bf16* Wn1p  = alloc(WN);
  __bf16* h0b   = alloc((size_t)B * D128);
  __bf16* mh1b  = alloc((size_t)B * D128);
  __bf16* h1b   = alloc((size_t)B * S2 * D128);
  __bf16* mh2b  = alloc((size_t)B * S2 * D128);
  __bf16* nh1b  = alloc((size_t)B * S2 * D128);
  __bf16* mnh1b = alloc((size_t)B * D128);
  __bf16* nh0b  = alloc((size_t)B * D128);

  // ---- pack weights into WMMA fragment layout ----
  sage_pack_w<<<WN / 256, 256, 0, stream>>>(W_self_0, Ws0p);
  sage_pack_w<<<WN / 256, 256, 0, stream>>>(W_neigh_0, Wn0p);
  sage_pack_w<<<WN / 256, 256, 0, stream>>>(W_self_1, Ws1p);
  sage_pack_w<<<WN / 256, 256, 0, stream>>>(W_neigh_1, Wn1p);

  // ---- gathers + neighbor means (bandwidth phase) ----
  sage_gather_bf16<<<B / 4, 128, 0, stream>>>(features, samples0, h0b);
  sage_gather_bf16<<<B * S2 / 4, 128, 0, stream>>>(features, samples1, h1b);
  sage_gather_mean_bf16<10><<<B / 4, 128, 0, stream>>>(features, samples1,
                                                       mh1b, 1.f / 10.f);
  sage_gather_mean_bf16<25><<<B * S2 / 4, 128, 0, stream>>>(features, samples2,
                                                            mh2b, 1.f / 25.f);

  // ---- layer 0 (relu) ----
  // nh1 = relu(h1 @ Ws0 + mean(h2) @ Wn0) : M=10240 -> 640 waves, 80 blocks
  sage_dual_gemm_wmma<true, false><<<B * S2 / 128, 256, 0, stream>>>(
      h1b, mh2b, Ws0p, Wn0p, nh1b, nullptr);
  // nh0 = relu(h0 @ Ws0 + mean(h1) @ Wn0) : M=1024 -> 64 waves, 8 blocks
  sage_dual_gemm_wmma<true, false><<<B / 128, 256, 0, stream>>>(
      h0b, mh1b, Ws0p, Wn0p, nh0b, nullptr);

  // ---- layer-1 aggregation ----
  sage_mean_rows_bf16<10><<<B / 4, 128, 0, stream>>>(nh1b, mnh1b, 1.f / 10.f);

  // ---- layer 1 (identity, f32 out) ----
  sage_dual_gemm_wmma<false, true><<<B / 128, 256, 0, stream>>>(
      nh0b, mnh1b, Ws1p, Wn1p, nullptr, out);
}
